// LlamaAttention_64080912056513
// MI455X (gfx1250) — matrix operand order repaired and checked by reading, not compiled, not run on hardware
//
#include <hip/hip_runtime.h>

// ---------------------------------------------------------------------------
// Types for CDNA5 WMMA / TDM (gfx1250, wave32)
// ---------------------------------------------------------------------------
typedef __attribute__((ext_vector_type(16))) __bf16          bf16x16;
typedef __attribute__((ext_vector_type(8)))  float           f32x8;
typedef __attribute__((ext_vector_type(8)))  unsigned short  u16x8;
typedef __attribute__((ext_vector_type(4)))  unsigned int    u32x4;
typedef __attribute__((ext_vector_type(8)))  int             i32x8;
typedef __attribute__((ext_vector_type(4)))  int             i32x4;

#define DEV static __device__ __forceinline__

DEV unsigned short f2bf(float f) {
  unsigned int u = __float_as_uint(f);
  u += 0x7fffu + ((u >> 16) & 1u);   // round-to-nearest-even
  return (unsigned short)(u >> 16);
}

DEV f32x8 zero8() { f32x8 z = {0.f,0.f,0.f,0.f,0.f,0.f,0.f,0.f}; return z; }

// A-matrix fragment (16x32 bf16). lanes 0-15: K=0..15, lanes 16-31:
// K=16..31. Caller pre-adds sel*16; we load p[0..15].
DEV bf16x16 frag_a(const unsigned short* p) {
  union { bf16x16 v; u16x8 u[2]; } r;
  r.u[0] = *(const u16x8*)(p);
  r.u[1] = *(const u16x8*)(p + 8);
  return r.v;
}
// B-matrix fragment (32x16 bf16). lanes 0-15: K=0..15, lanes 16-31: K=16..31.
// Caller pre-adds sel*16; we load p[0..15].
DEV bf16x16 frag_b(const unsigned short* p) {
  union { bf16x16 v; u16x8 u[2]; } r;
  r.u[0] = *(const u16x8*)(p);
  r.u[1] = *(const u16x8*)(p + 8);
  return r.v;
}

DEV f32x8 wmma_bf16(bf16x16 a, bf16x16 b, f32x8 c) {
  return __builtin_amdgcn_wmma_f32_16x16x32_bf16(false, a, false, b,
                                                 (short)0, c, false, false);
}

// ---------------------------------------------------------------------------
// Tensor Data Mover: async 2D tile (bf16) global -> LDS with row padding.
// Tile = tile_d1 rows x 32 halves; LDS row stride = 40 halves (pad 4 dwords
// every 16 dwords). OOB rows (beyond tensor_d1) are zero-filled by TDM.
// ---------------------------------------------------------------------------
DEV void tdm_load_tile(unsigned lds_addr, const unsigned short* gptr,
                       int tensor_d0, int tensor_d1, long long stride0,
                       unsigned tile_d1) {
  unsigned long long ga = (unsigned long long)(uintptr_t)gptr;
  u32x4 g0;
  g0[0] = 1u;                                         // count=1, no gather
  g0[1] = lds_addr;                                   // LDS byte address
  g0[2] = (unsigned)(ga & 0xffffffffu);               // global_addr[31:0]
  g0[3] = (unsigned)((ga >> 32) & 0x01ffffffu)        // global_addr[56:32]
        | (2u << 30);                                 // type=2 (image)
  unsigned td0 = (unsigned)tensor_d0;
  unsigned td1 = (unsigned)tensor_d1;
  unsigned s_lo = (unsigned)(stride0 & 0xffffffffll);
  unsigned s_hi = (unsigned)((stride0 >> 32) & 0xffffll);
  i32x8 g1;
  g1[0] = (int)((1u << 16)        // data_size = 2 bytes
              | (1u << 20)        // pad_enable
              | (3u << 22)        // pad_interval: 16 dwords
              | (3u << 25));      // pad_amount: 4 dwords  -> 40-half rows
  g1[1] = (int)((td0 & 0xffffu) << 16);               // tensor_dim0[15:0]
  g1[2] = (int)(((td0 >> 16) & 0xffffu)               // tensor_dim0[31:16]
              | ((td1 & 0xffffu) << 16));             // tensor_dim1[15:0]
  g1[3] = (int)(((td1 >> 16) & 0xffffu)               // tensor_dim1[31:16]
              | (32u << 16));                         // tile_dim0 = 32
  g1[4] = (int)(tile_d1 & 0xffffu);                   // tile_dim1; tile_dim2=0
  g1[5] = (int)s_lo;                                  // tensor_dim0_stride[31:0]
  g1[6] = (int)s_hi;                                  // stride[47:32]; dim1_stride lo = 0
  g1[7] = 0;
  i32x4 z4 = {0, 0, 0, 0};
#if defined(__clang_major__) && (__clang_major__ >= 23)
  i32x8 z8 = {0, 0, 0, 0, 0, 0, 0, 0};
  __builtin_amdgcn_tensor_load_to_lds(g0, g1, z4, z4, z8, 0);
#else
  __builtin_amdgcn_tensor_load_to_lds(g0, g1, z4, z4, 0);
#endif
}

// ---------------------------------------------------------------------------
// f32 -> bf16 conversion
// ---------------------------------------------------------------------------
__global__ __launch_bounds__(256)
void cvt_bf16_kernel(const float* __restrict__ src,
                     unsigned short* __restrict__ dst, long long n) {
  long long i = (long long)blockIdx.x * 1024 + (long long)threadIdx.x * 4;
  if (i + 3 < n) {
    float4 v = *(const float4*)(src + i);
    dst[i + 0] = f2bf(v.x); dst[i + 1] = f2bf(v.y);
    dst[i + 2] = f2bf(v.z); dst[i + 3] = f2bf(v.w);
  } else {
    for (int j = 0; j < 4; ++j)
      if (i + j < n) dst[i + j] = f2bf(src[i + j]);
  }
}

// ---------------------------------------------------------------------------
// Transpose + convert: src[R][C] f32 (row major) -> dst[C][R] bf16 (row major)
// ---------------------------------------------------------------------------
__global__ __launch_bounds__(256)
void transpose_cvt_kernel(const float* __restrict__ src,
                          unsigned short* __restrict__ dst, int R, int C) {
  __shared__ float tile[32][33];
  int c0 = blockIdx.x * 32, r0 = blockIdx.y * 32;
  int tx = threadIdx.x & 31, ty = threadIdx.x >> 5;   // 32 x 8
#pragma unroll
  for (int i = ty; i < 32; i += 8)
    tile[i][tx] = src[(size_t)(r0 + i) * C + c0 + tx];
  __syncthreads();
#pragma unroll
  for (int i = ty; i < 32; i += 8)
    dst[(size_t)(c0 + i) * R + r0 + tx] = f2bf(tile[tx][i]);
}

// ---------------------------------------------------------------------------
// Tiled bf16 GEMM with TDM double-buffered staging:
//   C[M,N] f32 = A[M,K] bf16 (row major) * BT[N,K] bf16 (row major, = B^T)
// Block tile 128x256, K-slice 32; 8 waves in a 2x4 grid, wave tile 64x64
// (16 WMMAs per slice against 16 ds_load_b128 -> 1.0 LDS loads per WMMA).
// Wave 0 DMAs both tiles for stage it+1, then waits stage it (tensorcnt<=2),
// so the Tensor Data Mover runs ahead of the WMMA pipeline.
// ---------------------------------------------------------------------------
__global__ __launch_bounds__(256)
void gemm_bf16_kernel(const unsigned short* __restrict__ A,
                      const unsigned short* __restrict__ BT,
                      float* __restrict__ C, int M, int N, int K) {
  __shared__ unsigned short sA[2][128][40];           // 20 KB
  __shared__ unsigned short sB[2][256][40];           // 40 KB

  int bm = blockIdx.y * 128, bn = blockIdx.x * 256;
  int tid  = threadIdx.x;
  int lane = tid & 31, wave = tid >> 5;
  int wm = (wave >> 2) * 64;                          // 0 or 64
  int wn = (wave & 3) * 64;                           // 0,64,128,192
  int colk = lane & 15;
  int selh = lane >> 4;

  unsigned aBase = (unsigned)(uintptr_t)(&sA[0][0][0]);
  unsigned bBase = (unsigned)(uintptr_t)(&sB[0][0][0]);
  const unsigned ATILE = 128u * 40u * 2u;             // 10240 bytes
  const unsigned BTILE = 256u * 40u * 2u;             // 20480 bytes
  int nk = K >> 5;

  f32x8 acc[4][4];
#pragma unroll
  for (int i = 0; i < 4; ++i)
#pragma unroll
    for (int j = 0; j < 4; ++j) acc[i][j] = zero8();

  if (wave == 0) {  // stage 0 into buffer 0
    tdm_load_tile(aBase, A  + (size_t)bm * K, K, M - bm, K, 128);
    tdm_load_tile(bBase, BT + (size_t)bn * K, K, N - bn, K, 256);
  }

  for (int it = 0; it < nk; ++it) {
    int cur = it & 1;
    if (wave == 0) {
      if (it + 1 < nk) {  // prefetch stage it+1 into the other buffer
        int kc = (it + 1) * 32;
        tdm_load_tile(aBase + (unsigned)(cur ^ 1) * ATILE,
                      A  + (size_t)bm * K + kc, K - kc, M - bm, K, 128);
        tdm_load_tile(bBase + (unsigned)(cur ^ 1) * BTILE,
                      BT + (size_t)bn * K + kc, K - kc, N - bn, K, 256);
        __builtin_amdgcn_s_wait_tensorcnt(2);  // stage it done, it+1 in flight
      } else {
        __builtin_amdgcn_s_wait_tensorcnt(0);
      }
    }
    __syncthreads();  // stage it visible to all waves

    const unsigned short* As = &sA[cur][0][0];
    const unsigned short* Bs = &sB[cur][0][0];
    bf16x16 bfrag[4];
#pragma unroll
    for (int j = 0; j < 4; ++j)
      bfrag[j] = frag_b(Bs + (size_t)(wn + j * 16 + colk) * 40 + selh * 16);
#pragma unroll
    for (int i = 0; i < 4; ++i) {
      bf16x16 a = frag_a(As + (size_t)(wm + i * 16 + colk) * 40 + selh * 16);
#pragma unroll
      for (int j = 0; j < 4; ++j)
        acc[i][j] = wmma_bf16(a, bfrag[j], acc[i][j]);
    }
    __syncthreads();  // all reads of buf[cur] done before TDM reuses it
  }

  int rofs = selh * 8;
#pragma unroll
  for (int i = 0; i < 4; ++i)
#pragma unroll
    for (int j = 0; j < 4; ++j)
#pragma unroll
      for (int r = 0; r < 8; ++r) {
        int row = bm + wm + i * 16 + rofs + r;
        if (row < M)
          C[(size_t)row * N + bn + wn + j * 16 + colk] = acc[i][j][r];
      }
}

// ---------------------------------------------------------------------------
// RoPE (in-place on f32 projections). pos = row % S for prefill rows, L for decode.
// ---------------------------------------------------------------------------
__global__ __launch_bounds__(256)
void rope_kernel(float* __restrict__ x, int T, int nheads, int S, int doff, int L) {
  const int D = 128, HALF = 64;
  long long idx   = (long long)blockIdx.x * 256 + threadIdx.x;
  long long total = (long long)T * nheads * HALF;
  if (idx >= total) return;
  int i   = (int)(idx % HALF);
  int h   = (int)((idx / HALF) % nheads);
  int row = (int)(idx / ((long long)HALF * nheads));
  int pos = row < doff ? (row % S) : L;
  float inv = __powf(10000.0f, -(float)(2 * i) / (float)D);
  float ang = (float)pos * inv;
  float sn, cn;
  __sincosf(ang, &sn, &cn);
  float* p = x + (size_t)row * nheads * D + h * D;
  float x1 = p[i], x2 = p[i + HALF];
  p[i]        = x1 * cn - x2 * sn;
  p[i + HALF] = x2 * cn + x1 * sn;
}

// ---------------------------------------------------------------------------
// Prefill flash attention: 1 wave per (batch, q-head, 16-query tile).
// QK^T and PV via v_wmma_f32_16x16x32_bf16; online softmax in VALU.
// ---------------------------------------------------------------------------
__global__ __launch_bounds__(32)
void prefill_attn_kernel(const unsigned short* __restrict__ qbf,  // [doff][4096] roped
                         const unsigned short* __restrict__ kbf,  // [doff][1024] roped
                         const unsigned short* __restrict__ vbf,  // [doff][1024]
                         float* __restrict__ attn) {              // [T][4096]
  const int S = 1024, Dq = 4096, Dkv = 1024;
  int q0  = blockIdx.x * 16;
  int hq  = blockIdx.y;
  int b   = blockIdx.z;
  int hkv = hq >> 2;                       // GQA group of 4
  int lane = threadIdx.x & 31;
  int colk = lane & 15;
  int selh = lane >> 4;

  __shared__ unsigned short Plds[16][40];  // P tile 16q x 32k (bf16)
  __shared__ unsigned short Vt[128][40];   // V tile transposed [d][key]

  const unsigned short* qrow =
      qbf + (size_t)(b * S + q0 + colk) * Dq + hq * 128 + selh * 16;
  bf16x16 aq[4];
#pragma unroll
  for (int c = 0; c < 4; ++c) aq[c] = frag_a(qrow + c * 32);

  f32x8 Of[8];
#pragma unroll
  for (int j = 0; j < 8; ++j) Of[j] = zero8();
  float m[8], l[8];
#pragma unroll
  for (int r = 0; r < 8; ++r) { m[r] = -1e30f; l[r] = 0.f; }

  const float scl = 0.08838834764831845f;  // 1/sqrt(128)
  int kend = q0 + 16;

  for (int kb = 0; kb < kend; kb += 32) {
    f32x8 s0 = zero8(), s1 = zero8();
    {
      int key0 = min(kb + colk, S - 1);
      int key1 = min(kb + 16 + colk, S - 1);
      const unsigned short* k0 =
          kbf + (size_t)(b * S + key0) * Dkv + hkv * 128 + selh * 16;
      const unsigned short* k1 =
          kbf + (size_t)(b * S + key1) * Dkv + hkv * 128 + selh * 16;
#pragma unroll
      for (int c = 0; c < 4; ++c) {
        s0 = wmma_bf16(aq[c], frag_b(k0 + c * 32), s0);
        s1 = wmma_bf16(aq[c], frag_b(k1 + c * 32), s1);
      }
    }
#pragma unroll
    for (int it = 0; it < 8; ++it) {
      int idx = it * 32 + lane;
      int kl = idx >> 3, seg = idx & 7;
      int key = min(kb + kl, S - 1);
      const unsigned short* vrow =
          vbf + (size_t)(b * S + key) * Dkv + hkv * 128 + seg * 16;
      u16x8 d0 = *(const u16x8*)vrow;
      u16x8 d1 = *(const u16x8*)(vrow + 8);
#pragma unroll
      for (int j2 = 0; j2 < 8; ++j2) Vt[seg * 16 + j2][kl]     = d0[j2];
#pragma unroll
      for (int j2 = 0; j2 < 8; ++j2) Vt[seg * 16 + 8 + j2][kl] = d1[j2];
    }
    float corr[8];
#pragma unroll
    for (int r = 0; r < 8; ++r) {
      int row = q0 + r + 8 * selh;
      float v0 = s0[r] * scl; if (kb + colk      > row) v0 = -1e30f;
      float v1 = s1[r] * scl; if (kb + 16 + colk > row) v1 = -1e30f;
      float mx = fmaxf(v0, v1);
#pragma unroll
      for (int o = 8; o; o >>= 1) mx = fmaxf(mx, __shfl_xor(mx, o, 32));
      float mn = fmaxf(m[r], mx);
      float p0 = __expf(v0 - mn);
      float p1 = __expf(v1 - mn);
      float rs = p0 + p1;
#pragma unroll
      for (int o = 8; o; o >>= 1) rs += __shfl_xor(rs, o, 32);
      corr[r] = __expf(m[r] - mn);
      l[r] = l[r] * corr[r] + rs;
      m[r] = mn;
      Plds[r + 8 * selh][colk]      = f2bf(p0);
      Plds[r + 8 * selh][16 + colk] = f2bf(p1);
    }
#pragma unroll
    for (int j = 0; j < 8; ++j)
#pragma unroll
      for (int r = 0; r < 8; ++r) Of[j][r] *= corr[r];
    __syncthreads();
    bf16x16 ap = frag_a(&Plds[colk][0] + selh * 16);
#pragma unroll
    for (int j = 0; j < 8; ++j) {
      bf16x16 bv = frag_b(&Vt[j * 16 + colk][0] + selh * 16);
      Of[j] = wmma_bf16(ap, bv, Of[j]);
    }
    __syncthreads();
  }
#pragma unroll
  for (int j = 0; j < 8; ++j)
#pragma unroll
    for (int r = 0; r < 8; ++r) {
      int row = q0 + r + 8 * selh;
      attn[(size_t)(b * S + row) * Dq + hq * 128 + j * 16 + colk] =
          Of[j][r] / l[r];
    }
}

// ---------------------------------------------------------------------------
// Decode: one block per (batch, kv-head); all 4 GQA q-heads per block so the
// f32 KV cache is streamed exactly once per kv-head.
// ---------------------------------------------------------------------------
__global__ __launch_bounds__(256)
void decode_attn_kernel(const float* __restrict__ qf,
                        const float* __restrict__ kf,
                        const float* __restrict__ vf,
                        const float* __restrict__ kcache,
                        const float* __restrict__ vcache,
                        float* __restrict__ attn) {
  const int L = 2048, doff = 4096, HKD = 8 * 128, D = 128, G = 4;
  int hkv = blockIdx.x, b = blockIdx.y;
  int tid = threadIdx.x;

  __shared__ float qsh[G][128];
  __shared__ float sc[G][L + 4];
  __shared__ float pacc[2][G][128];
  __shared__ float sM[G], sS[G];
  __shared__ float red[8];

  for (int i = tid; i < G * 128; i += 256) {
    int g = i >> 7, c = i & 127;
    qsh[g][c] = qf[(size_t)(doff + b) * 4096 + (hkv * G + g) * 128 + c];
  }
  __syncthreads();

  const float scl = 0.08838834764831845f;
  for (int k = tid; k <= L; k += 256) {
    const float* kp = (k < L)
        ? (kcache + ((size_t)b * L + k) * HKD + hkv * D)
        : (kf + (size_t)(doff + b) * 1024 + hkv * D);
    float s0 = 0, s1 = 0, s2 = 0, s3 = 0;
    for (int c = 0; c < D; ++c) {
      float kvv = kp[c];
      s0 += qsh[0][c] * kvv; s1 += qsh[1][c] * kvv;
      s2 += qsh[2][c] * kvv; s3 += qsh[3][c] * kvv;
    }
    sc[0][k] = s0 * scl; sc[1][k] = s1 * scl;
    sc[2][k] = s2 * scl; sc[3][k] = s3 * scl;
  }
  __syncthreads();

  for (int g = 0; g < G; ++g) {
    float lm = -1e30f;
    for (int k = tid; k <= L; k += 256) lm = fmaxf(lm, sc[g][k]);
    for (int o = 16; o; o >>= 1) lm = fmaxf(lm, __shfl_xor(lm, o, 32));
    if ((tid & 31) == 0) red[tid >> 5] = lm;
    __syncthreads();
    if (tid == 0) {
      float v = red[0];
      for (int w = 1; w < 8; ++w) v = fmaxf(v, red[w]);
      sM[g] = v;
    }
    __syncthreads();
    float mg = sM[g];
    float ls = 0.f;
    for (int k = tid; k <= L; k += 256) {
      float p = __expf(sc[g][k] - mg);
      sc[g][k] = p;
      ls += p;
    }
    for (int o = 16; o; o >>= 1) ls += __shfl_xor(ls, o, 32);
    if ((tid & 31) == 0) red[tid >> 5] = ls;
    __syncthreads();
    if (tid == 0) {
      float v = 0.f;
      for (int w = 0; w < 8; ++w) v += red[w];
      sS[g] = v;
    }
    __syncthreads();
  }

  int half = tid >> 7, col = tid & 127;
  int k0 = half ? 1025 : 0;
  int k1 = half ? (L + 1) : 1025;
  float a0 = 0, a1 = 0, a2 = 0, a3 = 0;
  for (int k = k0; k < k1; ++k) {
    const float* vp = (k < L)
        ? (vcache + ((size_t)b * L + k) * HKD + hkv * D + col)
        : (vf + (size_t)(doff + b) * 1024 + hkv * D + col);
    float vv = *vp;
    a0 += sc[0][k] * vv; a1 += sc[1][k] * vv;
    a2 += sc[2][k] * vv; a3 += sc[3][k] * vv;
  }
  pacc[half][0][col] = a0; pacc[half][1][col] = a1;
  pacc[half][2][col] = a2; pacc[half][3][col] = a3;
  __syncthreads();
  if (tid < 128) {
    for (int g = 0; g < G; ++g) {
      float o = (pacc[0][g][tid] + pacc[1][g][tid]) / sS[g];
      attn[(size_t)(doff + b) * 4096 + (hkv * G + g) * 128 + tid] = o;
    }
  }
}

// ---------------------------------------------------------------------------
// Host orchestration
// ---------------------------------------------------------------------------
extern "C" void kernel_launch(void* const* d_in, const int* in_sizes, int n_in,
                              void* d_out, int out_size, void* d_ws, size_t ws_size,
                              hipStream_t stream) {
  const float* hs     = (const float*)d_in[0];
  const float* wq     = (const float*)d_in[1];
  const float* wk     = (const float*)d_in[2];
  const float* wv     = (const float*)d_in[3];
  const float* wo     = (const float*)d_in[4];
  const float* kcache = (const float*)d_in[5];
  const float* vcache = (const float*)d_in[6];

  const int H = 4096, Hq = 32, Hkv = 8;
  const int Bp = 4, S = 1024, Bd = 32, L = 2048;
  const int doff = Bp * S, T = doff + Bd;         // 4096, 4128
  const int Dq = Hq * 128, Dkv = Hkv * 128;       // 4096, 1024

  char* ws = (char*)d_ws;
  size_t off = 0;
  auto alloc = [&](size_t bytes) {
    void* p = ws + off;
    off = (off + bytes + 255) & ~(size_t)255;
    return p;
  };
  unsigned short* hs_bf   = (unsigned short*)alloc((size_t)T * H * 2);
  unsigned short* wqT     = (unsigned short*)alloc((size_t)H * Dq * 2);   // [Dq][H]
  unsigned short* wkT     = (unsigned short*)alloc((size_t)H * Dkv * 2);  // [Dkv][H]
  unsigned short* wvT     = (unsigned short*)alloc((size_t)H * Dkv * 2);  // [Dkv][H]
  unsigned short* woT     = (unsigned short*)alloc((size_t)Dq * H * 2);   // [H][Dq]
  float*          q_proj  = (float*)alloc((size_t)T * Dq * 4);
  float*          k_proj  = (float*)alloc((size_t)T * Dkv * 4);
  float*          v_proj  = (float*)alloc((size_t)T * Dkv * 4);
  float*          attn    = (float*)alloc((size_t)T * Dq * 4);
  unsigned short* attn_bf = (unsigned short*)alloc((size_t)T * Dq * 2);
  unsigned short* qp_bf   = (unsigned short*)alloc((size_t)doff * Dq * 2);
  unsigned short* kp_bf   = (unsigned short*)alloc((size_t)doff * Dkv * 2);
  unsigned short* vp_bf   = (unsigned short*)alloc((size_t)doff * Dkv * 2);

  auto cvt = [&](const float* s, unsigned short* d, long long n) {
    cvt_bf16_kernel<<<dim3((unsigned)((n + 1023) / 1024)), dim3(256), 0, stream>>>(s, d, n);
  };
  auto tcvt = [&](const float* s, unsigned short* d, int R, int C) {
    transpose_cvt_kernel<<<dim3(C / 32, R / 32), dim3(256), 0, stream>>>(s, d, R, C);
  };
  auto gemm = [&](const unsigned short* A, const unsigned short* BT, float* C,
                  int M, int N, int K) {
    dim3 grid(N / 256, (M + 127) / 128);
    gemm_bf16_kernel<<<grid, dim3(256), 0, stream>>>(A, BT, C, M, N, K);
  };

  // 1) convert activations, transpose+convert weights (B^T for TDM GEMM)
  cvt(hs, hs_bf, (long long)T * H);
  tcvt(wq, wqT, H, Dq);
  tcvt(wk, wkT, H, Dkv);
  tcvt(wv, wvT, H, Dkv);
  tcvt(wo, woT, Dq, H);

  // 2) projections (TDM-staged WMMA GEMMs, f32 accumulate)
  gemm(hs_bf, wqT, q_proj, T, Dq, H);
  gemm(hs_bf, wkT, k_proj, T, Dkv, H);
  gemm(hs_bf, wvT, v_proj, T, Dkv, H);

  // 3) RoPE on q/k (in place, f32)
  {
    long long nq = (long long)T * Hq * 64;
    rope_kernel<<<dim3((unsigned)((nq + 255) / 256)), dim3(256), 0, stream>>>(
        q_proj, T, Hq, S, doff, L);
    long long nk = (long long)T * Hkv * 64;
    rope_kernel<<<dim3((unsigned)((nk + 255) / 256)), dim3(256), 0, stream>>>(
        k_proj, T, Hkv, S, doff, L);
  }

  // 4) bf16 copies of prefill q/k/v for WMMA attention
  cvt(q_proj, qp_bf, (long long)doff * Dq);
  cvt(k_proj, kp_bf, (long long)doff * Dkv);
  cvt(v_proj, vp_bf, (long long)doff * Dkv);

  // 5) prefill flash attention (WMMA)
  prefill_attn_kernel<<<dim3(S / 16, Hq, Bp), dim3(32), 0, stream>>>(
      qp_bf, kp_bf, vp_bf, attn);

  // 6) decode attention (VALU, cache streamed once per kv-head)
  decode_attn_kernel<<<dim3(Hkv, Bd), dim3(256), 0, stream>>>(
      q_proj, k_proj, v_proj, kcache, vcache, attn);

  // 7) output projection
  cvt(attn, attn_bf, (long long)T * Dq);
  gemm(attn_bf, woT, (float*)d_out, T, H, Dq);
}
